// SwinTransformerBlock_31988916421020
// MI455X (gfx1250) — matrix operand order repaired and checked by reading, not compiled, not run on hardware
//
#include <hip/hip_runtime.h>
#include <hip/hip_bf16.h>

// ---------------------------------------------------------------------------
// Types
// ---------------------------------------------------------------------------
typedef __bf16 bf16;
typedef __attribute__((ext_vector_type(16))) __bf16        v16bf;
typedef __attribute__((ext_vector_type(8)))  float         v8f;
typedef __attribute__((ext_vector_type(4)))  unsigned int  u32x4;
typedef __attribute__((ext_vector_type(4)))  float         f32x4;

__device__ __forceinline__ bf16 tobf(float f) { return (bf16)f; }

union FragU { u32x4 u[2]; v16bf v; };

// Generic -> LDS(addrspace 3) byte offset (wave-relative, as the ISA's
// async-load dsaddr wants).
typedef __attribute__((address_space(3))) const unsigned char lds_byte_t;
__device__ __forceinline__ unsigned lds_off(const void* p) {
  return (unsigned)(unsigned long long)(lds_byte_t*)p;
}

// Per-lane async DMA: 16 bytes global -> LDS, tracked by ASYNCcnt.
__device__ __forceinline__ void async_copy_b128(unsigned lds_byte_off,
                                                const void* gptr) {
  asm volatile("global_load_async_to_lds_b128 %0, %1, off"
               :: "v"(lds_byte_off), "v"(gptr)
               : "memory");
}

// A-fragment (16x32 bf16, MxK): lane holds row m=lane&15; K chunk is
// [k0 + half*16, +16)  (half = lane>>4).
__device__ __forceinline__ v16bf load_a_frag(const bf16* __restrict__ base,
                                             int ld, int row, int k0, int half) {
  const bf16* p = base + (size_t)row * ld + k0 + (half << 4);
  FragU f;
  f.u[0] = *(const u32x4*)p;
  f.u[1] = *(const u32x4*)(p + 8);
  return f.v;
}

// B-fragment (32x16 bf16, KxN) for NT gemm from K-major global storage.
__device__ __forceinline__ v16bf load_b_frag(const bf16* __restrict__ base,
                                             int ld, int col, int k0, int half) {
  const bf16* p = base + (size_t)col * ld + k0 + (half << 4);
  FragU f;
  f.u[0] = ((const u32x4*)p)[0];
  f.u[1] = ((const u32x4*)p)[1];
  return f.v;
}

#define WMMA_BF16(a, b, c) \
  __builtin_amdgcn_wmma_f32_16x16x32_bf16(false, (a), false, (b), (short)0, (c), false, false)

// ---------------------------------------------------------------------------
// Kernel: window partition + f32->bf16 cast.  row r = l*64 + n
// ---------------------------------------------------------------------------
__global__ __launch_bounds__(128) void winpart_kernel(
    const float* __restrict__ x, float* __restrict__ xw_f32,
    bf16* __restrict__ xw_bf16) {
  int r = blockIdx.x;
  int l = r >> 6, n = r & 63;
  int b = l >> 6, wh = (l >> 3) & 7, ww = l & 7;
  int i = n >> 3, j = n & 7;
  int p = (wh * 8 + i) * 64 + (ww * 8 + j);
  const f32x4* src = (const f32x4*)(x + ((size_t)b * 4096 + p) * 512);
  f32x4* dstf = (f32x4*)(xw_f32 + (size_t)r * 512);
  bf16* dstb = xw_bf16 + (size_t)r * 512;
  int t = threadIdx.x;  // 0..127 -> 4 floats each
  f32x4 v = src[t];
  dstf[t] = v;
  dstb[4 * t + 0] = tobf(v.x);
  dstb[4 * t + 1] = tobf(v.y);
  dstb[4 * t + 2] = tobf(v.z);
  dstb[4 * t + 3] = tobf(v.w);
}

// ---------------------------------------------------------------------------
// Kernel: f32 -> bf16 weight conversion
// ---------------------------------------------------------------------------
__global__ __launch_bounds__(256) void cvt_bf16_kernel(
    const float* __restrict__ in, bf16* __restrict__ out, int nElem) {
  int i = blockIdx.x * 256 + threadIdx.x;
  if (i < nElem) out[i] = tobf(in[i]);
}

// ---------------------------------------------------------------------------
// NT GEMM: C[M,N] = A[M,K] * B[N,K]^T + bias  (A,B bf16 K-contiguous)
// Block = 256 threads = 8 waves; tile 128x128; wave tile 32x64 (2x4 WMMAs).
// B tile (128x32 bf16 = 8KB) is staged to LDS with double-buffered
// global_load_async_to_lds_b128 (ASYNCcnt), overlapping DMA with WMMA.
// MODE 0: out_f32 = acc + bias + res_f32           (out-proj / FFN2 + residual)
// MODE 1: out_bf16 = gelu(acc + bias)              (FFN1)
// MODE 2: qkv scatter to q[n][h][l][d], k[n][h][l][d], v[n][h][d][l]
// ---------------------------------------------------------------------------
template <int MODE>
__global__ __launch_bounds__(256) void gemm_nt_kernel(
    const bf16* __restrict__ A, const bf16* __restrict__ B,
    const float* __restrict__ bias, int M, int N, int K,
    float* __restrict__ out_f32, const float* __restrict__ res_f32,
    bf16* __restrict__ out_bf16, bf16* __restrict__ qo,
    bf16* __restrict__ ko, bf16* __restrict__ vo) {
  __shared__ __align__(16) bf16 Bs[2][128 * 32];  // 16 KB double buffer

  const int tid = threadIdx.x;
  const int wave = tid >> 5, lane = tid & 31;
  const int half = lane >> 4, l16 = lane & 15;
  const int wm = wave & 3, wn = wave >> 2;
  const int bm = blockIdx.x * 128 + wm * 32;
  const int bn0 = blockIdx.y * 128;
  const int bn = bn0 + wn * 64;

  // Cooperative B-tile copy: 512 x 16B chunks, 2 per thread.
  // chunk i: col = i>>2 (0..127), part = i&3 (16B within the 64B row chunk).
  const int cc = tid >> 2;       // col for chunk tid (second chunk: col+64)
  const int pp = tid & 3;
  const unsigned bs_base = lds_off(&Bs[0][0]);
  auto issue_stage = [&](int k0, int buf) {
    const bf16* g0 = B + (size_t)(bn0 + cc) * K + k0 + pp * 8;
    const bf16* g1 = B + (size_t)(bn0 + 64 + cc) * K + k0 + pp * 8;
    unsigned l0 = bs_base + (unsigned)buf * 8192u + (unsigned)cc * 64u +
                  (unsigned)pp * 16u;
    async_copy_b128(l0, g0);
    async_copy_b128(l0 + 4096u, g1);
  };

  const int nstage = K >> 5;
  issue_stage(0, 0);

  v8f acc[2][4] = {};
  for (int s = 0; s < nstage; ++s) {
    const int k0 = s << 5;
    const int buf = s & 1;
    if (s + 1 < nstage) {
      issue_stage(k0 + 32, buf ^ 1);
      // 2 newest outstanding belong to stage s+1; oldest 2 (stage s) done.
      asm volatile("s_wait_asynccnt 0x2" ::: "memory");
    } else {
      asm volatile("s_wait_asynccnt 0x0" ::: "memory");
    }
    __syncthreads();  // publish stage-s LDS writes across all waves

    if (k0 + 32 < K)
      __builtin_prefetch(A + (size_t)(bm + l16) * K + k0 + 32, 0, 1);
    v16bf a0 = load_a_frag(A, K, bm + l16, k0, half);
    v16bf a1 = load_a_frag(A, K, bm + 16 + l16, k0, half);

    // B fragments from LDS: col c of tile at Bs[buf][c*32 + half*16]
    FragU fb[4];
#pragma unroll
    for (int j = 0; j < 4; ++j) {
      const bf16* p = &Bs[buf][(wn * 64 + j * 16 + l16) * 32 + (half << 4)];
      fb[j].u[0] = ((const u32x4*)p)[0];
      fb[j].u[1] = ((const u32x4*)p)[1];
    }
    acc[0][0] = WMMA_BF16(a0, fb[0].v, acc[0][0]);
    acc[0][1] = WMMA_BF16(a0, fb[1].v, acc[0][1]);
    acc[0][2] = WMMA_BF16(a0, fb[2].v, acc[0][2]);
    acc[0][3] = WMMA_BF16(a0, fb[3].v, acc[0][3]);
    acc[1][0] = WMMA_BF16(a1, fb[0].v, acc[1][0]);
    acc[1][1] = WMMA_BF16(a1, fb[1].v, acc[1][1]);
    acc[1][2] = WMMA_BF16(a1, fb[2].v, acc[1][2]);
    acc[1][3] = WMMA_BF16(a1, fb[3].v, acc[1][3]);

    __syncthreads();  // readers done before this buffer is refilled (s+2)
  }

  // Epilogue. C layout: value acc[i][j][r] is C[bm+i*16 + r + 8*half, bn+j*16+l16]
#pragma unroll
  for (int i = 0; i < 2; ++i) {
#pragma unroll
    for (int j = 0; j < 4; ++j) {
#pragma unroll
      for (int r = 0; r < 8; ++r) {
        int row = bm + i * 16 + r + 8 * half;
        int col = bn + j * 16 + l16;
        float v = acc[i][j][r] + bias[col];
        if constexpr (MODE == 0) {
          size_t o = (size_t)row * N + col;
          out_f32[o] = v + res_f32[o];
        } else if constexpr (MODE == 1) {
          float g = 0.5f * v * (1.0f + erff(v * 0.70710678118654752f));
          out_bf16[(size_t)row * N + col] = tobf(g);
        } else {  // MODE 2: QKV scatter; row = l*64 + n
          int l = row >> 6, n = row & 63;
          if (col < 512) {
            int h = col >> 5, d = col & 31;
            qo[(((size_t)n * 16 + h) * 256 + l) * 32 + d] = tobf(v);
          } else if (col < 1024) {
            int c = col - 512, h = c >> 5, d = c & 31;
            ko[(((size_t)n * 16 + h) * 256 + l) * 32 + d] = tobf(v);
          } else {
            int c = col - 1024, h = c >> 5, d = c & 31;
            vo[(((size_t)n * 16 + h) * 32 + d) * 256 + l] = tobf(v);
          }
        }
      }
    }
  }
}

// ---------------------------------------------------------------------------
// Attention: per (n,h) head, per 32-row strip of L=256.
// S = Q K^T * scale (WMMA, K=32), softmax in LDS, O = P V (WMMA over m).
// Block = 64 threads (2 waves); each wave owns 16 rows.
// ---------------------------------------------------------------------------
__global__ __launch_bounds__(64) void attn_kernel(
    const bf16* __restrict__ qb, const bf16* __restrict__ kb,
    const bf16* __restrict__ vb, bf16* __restrict__ ob) {
  __shared__ float S[32 * 257];  // stride 257 -> conflict-free row scans
  const int nh = blockIdx.x;     // n*16 + h
  const int n = nh >> 4, h = nh & 15;
  const int strip = blockIdx.y;  // 0..7
  const int tid = threadIdx.x;
  const int wave = tid >> 5, lane = tid & 31;
  const int half = lane >> 4, l16 = lane & 15;
  const float scale = 0.17677669529663687f;  // 1/sqrt(32)

  const bf16* Qn = qb + (size_t)nh * 256 * 32;
  const bf16* Kn = kb + (size_t)nh * 256 * 32;
  const bf16* Vn = vb + (size_t)nh * 32 * 256;
  const int rbase = strip * 32 + wave * 16;  // first l-row of this wave tile

  // Phase 1: scores strip 16x256 per wave, one WMMA (k=d=32) per 16-col tile.
  v16bf aq = load_a_frag(Qn, 32, rbase + l16, 0, half);
  for (int ct = 0; ct < 16; ++ct) {
    v16bf bk = load_b_frag(Kn, 32, ct * 16 + l16, 0, half);
    v8f sc = {};
    sc = WMMA_BF16(aq, bk, sc);
#pragma unroll
    for (int r = 0; r < 8; ++r)
      S[(wave * 16 + r + 8 * half) * 257 + ct * 16 + l16] = sc[r] * scale;
  }
  __syncthreads();

  // Phase 2: row softmax (threads 0..31, one row each)
  if (tid < 32) {
    float* row = &S[tid * 257];
    float m = -1e30f;
    for (int c = 0; c < 256; ++c) m = fmaxf(m, row[c]);
    float s = 0.0f;
    for (int c = 0; c < 256; ++c) {
      float p = __expf(row[c] - m);
      row[c] = p;
      s += p;
    }
    float inv = 1.0f / s;
    for (int c = 0; c < 256; ++c) row[c] *= inv;
  }
  __syncthreads();

  // Phase 3: O(16x32) = P(16x256) @ V(256x32); V stored transposed [d][m] (NT).
  v8f oacc[2] = {};
  for (int mt = 0; mt < 256; mt += 32) {
    union { v16bf v; bf16 e[16]; } af;
    const float* srow = &S[(wave * 16 + l16) * 257 + mt];
#pragma unroll
    for (int q = 0; q < 8; ++q) af.e[q] = tobf(srow[half * 16 + q]);
#pragma unroll
    for (int q = 0; q < 8; ++q) af.e[8 + q] = tobf(srow[half * 16 + 8 + q]);
#pragma unroll
    for (int j = 0; j < 2; ++j) {
      v16bf bv = load_b_frag(Vn, 256, j * 16 + l16, mt, half);
      oacc[j] = WMMA_BF16(af.v, bv, oacc[j]);
    }
  }
#pragma unroll
  for (int j = 0; j < 2; ++j) {
#pragma unroll
    for (int r = 0; r < 8; ++r) {
      int l = rbase + r + 8 * half;
      int col = h * 32 + j * 16 + l16;
      ob[((size_t)l * 64 + n) * 512 + col] = tobf(oacc[j][r]);
    }
  }
}

// ---------------------------------------------------------------------------
// LayerNorm over 512; MAP_OUT=false -> writes y_f32 + y_bf16;
// MAP_OUT=true -> window-unpartition and write f32 to d_out.
// ---------------------------------------------------------------------------
template <bool MAP_OUT>
__global__ __launch_bounds__(128) void layernorm_kernel(
    const float* __restrict__ zin, const float* __restrict__ g,
    const float* __restrict__ bt, float* __restrict__ y_f32,
    bf16* __restrict__ y_bf16, float* __restrict__ mapped_out) {
  __shared__ float red1[128];
  __shared__ float red2[128];
  int r = blockIdx.x;
  int t = threadIdx.x;
  f32x4 v = ((const f32x4*)(zin + (size_t)r * 512))[t];
  red1[t] = v.x + v.y + v.z + v.w;
  red2[t] = v.x * v.x + v.y * v.y + v.z * v.z + v.w * v.w;
  __syncthreads();
  for (int off = 64; off > 0; off >>= 1) {
    if (t < off) {
      red1[t] += red1[t + off];
      red2[t] += red2[t + off];
    }
    __syncthreads();
  }
  float mu = red1[0] * (1.0f / 512.0f);
  float var = red2[0] * (1.0f / 512.0f) - mu * mu;
  float rstd = rsqrtf(var + 1e-5f);
  f32x4 gg = ((const f32x4*)g)[t];
  f32x4 bb = ((const f32x4*)bt)[t];
  f32x4 o;
  o.x = (v.x - mu) * rstd * gg.x + bb.x;
  o.y = (v.y - mu) * rstd * gg.y + bb.y;
  o.z = (v.z - mu) * rstd * gg.z + bb.z;
  o.w = (v.w - mu) * rstd * gg.w + bb.w;
  if constexpr (MAP_OUT) {
    int l = r >> 6, n = r & 63;
    int b = l >> 6, wh = (l >> 3) & 7, ww = l & 7;
    int i = n >> 3, j = n & 7;
    int p = (wh * 8 + i) * 64 + (ww * 8 + j);
    ((f32x4*)(mapped_out + ((size_t)b * 4096 + p) * 512))[t] = o;
  } else {
    ((f32x4*)(y_f32 + (size_t)r * 512))[t] = o;
    bf16* d = y_bf16 + (size_t)r * 512 + 4 * t;
    d[0] = tobf(o.x);
    d[1] = tobf(o.y);
    d[2] = tobf(o.z);
    d[3] = tobf(o.w);
  }
}

// ---------------------------------------------------------------------------
// Host launcher
// ---------------------------------------------------------------------------
extern "C" void kernel_launch(void* const* d_in, const int* in_sizes, int n_in,
                              void* d_out, int out_size, void* d_ws,
                              size_t ws_size, hipStream_t stream) {
  (void)in_sizes; (void)n_in; (void)out_size; (void)ws_size;
  const float* x         = (const float*)d_in[0];
  const float* in_proj_w = (const float*)d_in[1];
  const float* in_proj_b = (const float*)d_in[2];
  const float* out_w     = (const float*)d_in[3];
  const float* out_b     = (const float*)d_in[4];
  const float* ln1_g     = (const float*)d_in[5];
  const float* ln1_b     = (const float*)d_in[6];
  const float* w1        = (const float*)d_in[7];
  const float* b1        = (const float*)d_in[8];
  const float* w2        = (const float*)d_in[9];
  const float* b2        = (const float*)d_in[10];
  const float* ln2_g     = (const float*)d_in[11];
  const float* ln2_b     = (const float*)d_in[12];

  const int M = 16384;  // L*N = 256 windows * 64 tokens

  char* ws = (char*)d_ws;
  size_t off = 0;
  auto alloc = [&](size_t bytes) {
    size_t o = off;
    off += (bytes + 255) & ~(size_t)255;
    return o;
  };
  size_t o_xwb = alloc((size_t)M * 512 * 2);       // xw bf16
  size_t o_xwf = alloc((size_t)M * 512 * 4);       // xw f32 (residual 1) / z2 alias
  size_t o_wq  = alloc((size_t)1536 * 512 * 2);    // in_proj bf16
  size_t o_wo  = alloc((size_t)512 * 512 * 2);     // out_w bf16
  size_t o_w1  = alloc((size_t)2048 * 512 * 2);    // w1 bf16
  size_t o_w2  = alloc((size_t)512 * 2048 * 2);    // w2 bf16
  size_t o_qb  = alloc((size_t)16 * 64 * 256 * 32 * 2);  // q [n][h][l][d]
  size_t o_kb  = alloc((size_t)16 * 64 * 256 * 32 * 2);  // k
  size_t o_vb  = alloc((size_t)16 * 64 * 32 * 256 * 2);  // v transposed
  size_t o_ob  = alloc((size_t)M * 512 * 2);       // attn context bf16
  size_t o_z1  = alloc((size_t)M * 512 * 4);       // resid1 f32
  size_t o_yf  = alloc((size_t)M * 512 * 4);       // y f32
  size_t o_yb  = alloc((size_t)M * 512 * 2);       // y bf16
  size_t o_h1  = o_qb;                             // FFN hidden bf16 aliases q/k/v/o (64MB)
  size_t o_z2  = o_xwf;                            // resid2 f32 aliases xw_f32

  bf16*  xwb = (bf16*)(ws + o_xwb);
  float* xwf = (float*)(ws + o_xwf);
  bf16*  wq  = (bf16*)(ws + o_wq);
  bf16*  wo  = (bf16*)(ws + o_wo);
  bf16*  w1b = (bf16*)(ws + o_w1);
  bf16*  w2b = (bf16*)(ws + o_w2);
  bf16*  qb  = (bf16*)(ws + o_qb);
  bf16*  kb  = (bf16*)(ws + o_kb);
  bf16*  vb  = (bf16*)(ws + o_vb);
  bf16*  ob  = (bf16*)(ws + o_ob);
  float* z1  = (float*)(ws + o_z1);
  float* yf  = (float*)(ws + o_yf);
  bf16*  yb  = (bf16*)(ws + o_yb);
  bf16*  h1  = (bf16*)(ws + o_h1);
  float* z2  = (float*)(ws + o_z2);

  // 1. window partition + cast
  winpart_kernel<<<M, 128, 0, stream>>>(x, xwf, xwb);

  // 2. weight casts
  cvt_bf16_kernel<<<(1536 * 512 + 255) / 256, 256, 0, stream>>>(in_proj_w, wq, 1536 * 512);
  cvt_bf16_kernel<<<(512 * 512 + 255) / 256, 256, 0, stream>>>(out_w, wo, 512 * 512);
  cvt_bf16_kernel<<<(2048 * 512 + 255) / 256, 256, 0, stream>>>(w1, w1b, 2048 * 512);
  cvt_bf16_kernel<<<(512 * 2048 + 255) / 256, 256, 0, stream>>>(w2, w2b, 512 * 2048);

  // 3. QKV projection with head-major scatter
  gemm_nt_kernel<2><<<dim3(M / 128, 1536 / 128), 256, 0, stream>>>(
      xwb, wq, in_proj_b, M, 1536, 512, nullptr, nullptr, nullptr, qb, kb, vb);

  // 4. attention per (n,h) x 8 row strips
  attn_kernel<<<dim3(64 * 16, 8), 64, 0, stream>>>(qb, kb, vb, ob);

  // 5. out projection + residual -> z1
  gemm_nt_kernel<0><<<dim3(M / 128, 512 / 128), 256, 0, stream>>>(
      ob, wo, out_b, M, 512, 512, z1, xwf, nullptr, nullptr, nullptr, nullptr);

  // 6. LN1 -> y (f32 + bf16)
  layernorm_kernel<false><<<M, 128, 0, stream>>>(z1, ln1_g, ln1_b, yf, yb, nullptr);

  // 7. FFN1 + exact GELU -> h1 bf16
  gemm_nt_kernel<1><<<dim3(M / 128, 2048 / 128), 256, 0, stream>>>(
      yb, w1b, b1, M, 2048, 512, nullptr, nullptr, h1, nullptr, nullptr, nullptr);

  // 8. FFN2 + residual -> z2
  gemm_nt_kernel<0><<<dim3(M / 128, 512 / 128), 256, 0, stream>>>(
      h1, w2b, b2, M, 512, 2048, z2, yf, nullptr, nullptr, nullptr, nullptr);

  // 9. LN2 + window unpartition -> d_out (f32)
  layernorm_kernel<true><<<M, 128, 0, stream>>>(z2, ln2_g, ln2_b, nullptr, nullptr,
                                                (float*)d_out);
}
